// PointNetPP_UpModule_90185723281829
// MI455X (gfx1250) — matrix operand order repaired and checked by reading, not compiled, not run on hardware
//
#include <hip/hip_runtime.h>

// ---------------------------------------------------------------------------
// PointNet++ feature-propagation (upsampling) module for MI455X (gfx1250).
//  1) 3-NN inverse-distance interpolation (LDS-staged point cloud, f32 VALU)
//  2) concat -> [1x1 conv + BN + ReLU] x2 via f16 WMMA GEMMs (f32 accum),
//     tiles staged with CDNA5 async global->LDS b128 (ASYNCcnt) copies
//  3) deterministic BN stats (two-stage column reduction, no atomics)
// ---------------------------------------------------------------------------

typedef _Float16 half_t;
typedef __attribute__((ext_vector_type(16))) _Float16 v16h;
typedef __attribute__((ext_vector_type(8)))  float    v8f;
typedef __attribute__((ext_vector_type(4)))  _Float16 h4v;
typedef __attribute__((ext_vector_type(4)))  float    f4v;

union Frag16 { unsigned int u[8]; v16h h; };

#define B_    16
#define N1_   1024
#define N2_   4096
#define C_    256
#define CP_   128
#define KIN_  384      // C_ + CP_
#define NOUT_ 256
#define M_    65536    // B_*N2_

// 32-bit LDS offset of a shared-memory pointer (addrspace(3) is 32-bit).
__device__ __forceinline__ unsigned lds_off(const void* p) {
  return (unsigned)(unsigned long long)(const __attribute__((address_space(3))) void*)p;
}

// CDNA5 async global->LDS 128-bit copy (tracked by ASYNCcnt).
__device__ __forceinline__ void async_copy_b128(unsigned lds_byte_off,
                                                const void* gaddr) {
  asm volatile("global_load_async_to_lds_b128 %0, %1, off"
               :: "v"(lds_byte_off),
                  "v"((unsigned long long)(unsigned long long)gaddr)
               : "memory");
}
__device__ __forceinline__ void wait_async0() {
  asm volatile("s_wait_asynccnt 0x0" ::: "memory");
}

// ---------------------------- 3-NN search ---------------------------------
__global__ __launch_bounds__(256) void knn_kernel(
    const float* __restrict__ pos,       // [B,N1,3]
    const float* __restrict__ prev_pos,  // [B,N2,3]
    int* __restrict__ oidx,              // [M,3]
    float* __restrict__ ow)              // [M,3]
{
  __shared__ __align__(16) float sp[N1_ * 3];   // 12 KB
  const int b     = blockIdx.x >> 4;   // 16 chunks per batch
  const int chunk = blockIdx.x & 15;
  const f4v* p1v = (const f4v*)(pos + (size_t)b * N1_ * 3);
  for (int i = threadIdx.x; i < (N1_ * 3) / 4; i += 256)
    ((f4v*)sp)[i] = p1v[i];
  __syncthreads();

  const int n2 = chunk * 256 + threadIdx.x;
  const size_t q = (size_t)b * N2_ + n2;
  const float px = prev_pos[q * 3 + 0];
  const float py = prev_pos[q * 3 + 1];
  const float pz = prev_pos[q * 3 + 2];

  float d0 = 1e30f, d1 = 1e30f, d2 = 1e30f;
  int   i0 = 0,     i1 = 0,     i2 = 0;
  for (int j = 0; j < N1_; ++j) {
    const float dx = px - sp[j * 3 + 0];
    const float dy = py - sp[j * 3 + 1];
    const float dz = pz - sp[j * 3 + 2];
    const float d = dx * dx + dy * dy + dz * dz;
    if (d < d0)       { d2 = d1; i2 = i1; d1 = d0; i1 = i0; d0 = d; i0 = j; }
    else if (d < d1)  { d2 = d1; i2 = i1; d1 = d;  i1 = j; }
    else if (d < d2)  { d2 = d;  i2 = j; }
  }
  const float r0 = 1.f / (sqrtf(d0) + 1e-8f);
  const float r1 = 1.f / (sqrtf(d1) + 1e-8f);
  const float r2 = 1.f / (sqrtf(d2) + 1e-8f);
  const float rs = 1.f / (r0 + r1 + r2);
  oidx[q * 3 + 0] = i0; oidx[q * 3 + 1] = i1; oidx[q * 3 + 2] = i2;
  ow[q * 3 + 0] = r0 * rs; ow[q * 3 + 1] = r1 * rs; ow[q * 3 + 2] = r2 * rs;
}

// ----------------- gather + interpolate + concat -> f16 A ------------------
// Vectorized: one float4 of channels per iteration.
__global__ __launch_bounds__(256) void build_h_kernel(
    const float* __restrict__ x,       // [B,N1,C]
    const float* __restrict__ prev_x,  // [B,N2,CP]
    const int* __restrict__ idx,
    const float* __restrict__ w,
    half_t* __restrict__ h)            // [M, KIN_] f16 row-major
{
  const f4v* xv  = (const f4v*)x;        // row stride C_/4 = 64
  const f4v* pv  = (const f4v*)prev_x;   // row stride CP_/4 = 32
  h4v* hv = (h4v*)h;                     // row stride KIN_/4 = 96
  const long long tot = (long long)M_ * (KIN_ / 4);
  for (long long e = (long long)blockIdx.x * blockDim.x + threadIdx.x; e < tot;
       e += (long long)gridDim.x * blockDim.x) {
    const int m  = (int)(e / (KIN_ / 4));
    const int k4 = (int)(e - (long long)m * (KIN_ / 4));
    f4v v;
    if (k4 < C_ / 4) {
      const int b = m >> 12;                                    // m / N2_
      const f4v* xb = xv + (size_t)b * N1_ * (C_ / 4);
      const int j0 = idx[(size_t)m * 3 + 0];
      const int j1 = idx[(size_t)m * 3 + 1];
      const int j2 = idx[(size_t)m * 3 + 2];
      const float w0 = w[(size_t)m * 3 + 0];
      const float w1 = w[(size_t)m * 3 + 1];
      const float w2 = w[(size_t)m * 3 + 2];
      v = w0 * xb[(size_t)j0 * (C_ / 4) + k4]
        + w1 * xb[(size_t)j1 * (C_ / 4) + k4]
        + w2 * xb[(size_t)j2 * (C_ / 4) + k4];
    } else {
      v = pv[(size_t)m * (CP_ / 4) + (k4 - C_ / 4)];
    }
    hv[e] = __builtin_convertvector(v, h4v);
  }
}

// --------------------------- f32 -> f16 convert ----------------------------
__global__ void cvt_f16_kernel(const float* __restrict__ src,
                               half_t* __restrict__ dst, int n4) {
  const int i = blockIdx.x * blockDim.x + threadIdx.x;
  if (i < n4)
    ((h4v*)dst)[i] = __builtin_convertvector(((const f4v*)src)[i], h4v);
}

// ------------------ WMMA GEMM: Z[m,n] = A[m,:]·W[n,:] + b[n] ---------------
// A: [M, K] f16 row-major; W: [NOUT_, K] f16 row-major; Z: [M, NOUT_] f32.
// Block: 256 thr (8 waves), tile 128(M) x 128(N), BK=32.
// Waves 4(M) x 2(N); each wave owns 32x64 = 2x4 WMMA 16x16 tiles.
// Tiles staged via async global->LDS b128; LDS row stride 20 uints (80 B,
// 16B-aligned chunks, 16 distinct banks across a fragment's 16 lanes).
#define BM 128
#define BN 128
#define BK 32
#define LSTR 20
__global__ __launch_bounds__(256) void gemm_kernel(
    const half_t* __restrict__ A, const half_t* __restrict__ W,
    const float* __restrict__ bias, float* __restrict__ Z, int K)
{
  __shared__ __align__(16) unsigned int la[BM][LSTR];
  __shared__ __align__(16) unsigned int lb[BN][LSTR];

  const int tid  = threadIdx.x;
  const int lane = tid & 31;
  const int wave = tid >> 5;
  const int wm   = wave & 3;        // wave M index (0..3)
  const int wn   = wave >> 2;       // wave N index (0..1)
  const int ln15 = lane & 15;
  const int half = lane >> 4;
  const int mBase = blockIdx.x * BM;
  const int nBase = blockIdx.y * BN;

  const unsigned int* Au = (const unsigned int*)A;
  const unsigned int* Wu = (const unsigned int*)W;
  const int Ku = K >> 1;            // uints (f16 pairs) per row

  const unsigned laBase = lds_off(&la[0][0]);
  const unsigned lbBase = lds_off(&lb[0][0]);

  // Per-thread staging chunks: 512 uint4 per tile / 256 thr = 2 each.
  const int c0 = tid, c1 = tid + 256;
  const int r0 = c0 >> 2, q0 = c0 & 3;
  const int r1 = c1 >> 2, q1 = c1 & 3;

  v8f acc[2][4] = {};

  for (int kb = 0; kb < K; kb += BK) {
    const int kbu = kb >> 1;
    __syncthreads();                       // LDS safe to overwrite
    async_copy_b128(laBase + (unsigned)(r0 * LSTR + q0 * 4) * 4,
                    Au + (size_t)(mBase + r0) * Ku + kbu + q0 * 4);
    async_copy_b128(laBase + (unsigned)(r1 * LSTR + q1 * 4) * 4,
                    Au + (size_t)(mBase + r1) * Ku + kbu + q1 * 4);
    async_copy_b128(lbBase + (unsigned)(r0 * LSTR + q0 * 4) * 4,
                    Wu + (size_t)(nBase + r0) * Ku + kbu + q0 * 4);
    async_copy_b128(lbBase + (unsigned)(r1 * LSTR + q1 * 4) * 4,
                    Wu + (size_t)(nBase + r1) * Ku + kbu + q1 * 4);
    wait_async0();                         // my async writes done
    __syncthreads();                       // everyone's writes visible

    Frag16 af[2], bf[4];
#pragma unroll
    for (int s = 0; s < 2; ++s) {          // A 16x32 f16 layout (ISA 7.12.2)
      const int m = wm * 32 + s * 16 + ln15;
#pragma unroll
      for (int v = 0; v < 8; ++v) {
        const int kp = half * 4 + (v & 3) + ((v >> 2) << 3);
        af[s].u[v] = la[m][kp];
      }
    }
#pragma unroll
    for (int j = 0; j < 4; ++j) {          // B 32x16 f16 layout
      const int n = wn * 64 + j * 16 + ln15;
#pragma unroll
      for (int v = 0; v < 8; ++v) {
        const int kp = half * 4 + (v & 3) + ((v >> 2) << 3);
        bf[j].u[v] = lb[n][kp];
      }
    }
#pragma unroll
    for (int i = 0; i < 2; ++i)
#pragma unroll
      for (int j = 0; j < 4; ++j)
        acc[i][j] = __builtin_amdgcn_wmma_f32_16x16x32_f16(
            false, af[i].h, false, bf[j].h, (short)0, acc[i][j], false, false);
  }

  // Epilogue: bias add + store (C/D layout: col = lane&15, row = r + 8*half)
#pragma unroll
  for (int i = 0; i < 2; ++i) {
#pragma unroll
    for (int j = 0; j < 4; ++j) {
      const int n = nBase + wn * 64 + j * 16 + ln15;
      const float bb = bias[n];
#pragma unroll
      for (int r = 0; r < 8; ++r) {
        const int m = mBase + wm * 32 + i * 16 + r + half * 8;
        Z[(size_t)m * NOUT_ + n] = acc[i][j][r] + bb;
      }
    }
  }
}

// ------------- deterministic BN stats: stage 1 (row-chunk sums) ------------
// 64 threads, each owns 4 channels (one float4); fully coalesced b128 loads.
__global__ __launch_bounds__(64) void colsum_kernel(
    const float* __restrict__ z, float* __restrict__ psum, float* __restrict__ psq)
{
  const int o4 = threadIdx.x;         // channel/4
  const int p  = blockIdx.x;          // row chunk (256 rows)
  const f4v* zp = (const f4v*)z + (size_t)p * 256 * (NOUT_ / 4);
  f4v s = {0.f, 0.f, 0.f, 0.f};
  f4v q = {0.f, 0.f, 0.f, 0.f};
  for (int r = 0; r < 256; ++r) {
    const f4v v = zp[(size_t)r * (NOUT_ / 4) + o4];
    s += v; q += v * v;
  }
  ((f4v*)psum)[(size_t)p * (NOUT_ / 4) + o4] = s;
  ((f4v*)psq)[(size_t)p * (NOUT_ / 4) + o4] = q;
}

// ------------- stage 2: finalize mu / rsqrt(var+eps) per channel -----------
__global__ void bn_stats_kernel(const float* __restrict__ psum,
                                const float* __restrict__ psq,
                                float* __restrict__ mu, float* __restrict__ rsig)
{
  const int o = threadIdx.x;
  float s = 0.f, q = 0.f;
  for (int p = 0; p < 256; ++p) {
    s += psum[(size_t)p * NOUT_ + o];
    q += psq[(size_t)p * NOUT_ + o];
  }
  const float m = s * (1.0f / (float)M_);
  const float v = q * (1.0f / (float)M_) - m * m;
  mu[o]   = m;
  rsig[o] = rsqrtf(v + 1e-5f);
}

// --------------------- normalize + ReLU (-> f16 or f32) --------------------
__global__ __launch_bounds__(256) void norm_relu_f16_kernel(
    const float* __restrict__ z, const float* __restrict__ mu,
    const float* __restrict__ rsig, const float* __restrict__ g,
    const float* __restrict__ be, half_t* __restrict__ out)
{
  const f4v* zv = (const f4v*)z;
  const long long tot = (long long)M_ * (NOUT_ / 4);
  for (long long e = (long long)blockIdx.x * blockDim.x + threadIdx.x; e < tot;
       e += (long long)gridDim.x * blockDim.x) {
    const int o4 = (int)(e & (NOUT_ / 4 - 1));
    f4v v = (zv[e] - ((const f4v*)mu)[o4]) * ((const f4v*)rsig)[o4] *
                ((const f4v*)g)[o4] + ((const f4v*)be)[o4];
    v.x = fmaxf(v.x, 0.f); v.y = fmaxf(v.y, 0.f);
    v.z = fmaxf(v.z, 0.f); v.w = fmaxf(v.w, 0.f);
    ((h4v*)out)[e] = __builtin_convertvector(v, h4v);
  }
}

__global__ __launch_bounds__(256) void norm_relu_f32_kernel(
    const float* __restrict__ z, const float* __restrict__ mu,
    const float* __restrict__ rsig, const float* __restrict__ g,
    const float* __restrict__ be, float* __restrict__ out)
{
  const f4v* zv = (const f4v*)z;
  const long long tot = (long long)M_ * (NOUT_ / 4);
  for (long long e = (long long)blockIdx.x * blockDim.x + threadIdx.x; e < tot;
       e += (long long)gridDim.x * blockDim.x) {
    const int o4 = (int)(e & (NOUT_ / 4 - 1));
    f4v v = (zv[e] - ((const f4v*)mu)[o4]) * ((const f4v*)rsig)[o4] *
                ((const f4v*)g)[o4] + ((const f4v*)be)[o4];
    v.x = fmaxf(v.x, 0.f); v.y = fmaxf(v.y, 0.f);
    v.z = fmaxf(v.z, 0.f); v.w = fmaxf(v.w, 0.f);
    ((f4v*)out)[e] = v;
  }
}

// ------------------------------- f32 copy ----------------------------------
__global__ void copy_f32x4_kernel(const float* __restrict__ src,
                                  float* __restrict__ dst, int n4) {
  const int i = blockIdx.x * blockDim.x + threadIdx.x;
  if (i < n4) ((f4v*)dst)[i] = ((const f4v*)src)[i];
}

// ---------------------------------------------------------------------------
extern "C" void kernel_launch(void* const* d_in, const int* in_sizes, int n_in,
                              void* d_out, int out_size, void* d_ws, size_t ws_size,
                              hipStream_t stream) {
  const float* x        = (const float*)d_in[0];
  const float* pos      = (const float*)d_in[1];
  const float* prev_x   = (const float*)d_in[2];
  const float* prev_pos = (const float*)d_in[3];
  const float* W0 = (const float*)d_in[4];
  const float* b0 = (const float*)d_in[5];
  const float* g0 = (const float*)d_in[6];
  const float* be0 = (const float*)d_in[7];
  const float* W1 = (const float*)d_in[8];
  const float* b1 = (const float*)d_in[9];
  const float* g1 = (const float*)d_in[10];
  const float* be1 = (const float*)d_in[11];

  float* out = (float*)d_out;

  // ---- workspace layout (all 256B aligned) ----
  char* ws = (char*)d_ws;
  size_t off = 0;
  auto alloc = [&](size_t bytes) {
    void* p = ws + off;
    off += (bytes + 255) & ~(size_t)255;
    return p;
  };
  int*    idx  = (int*)   alloc((size_t)M_ * 3 * sizeof(int));
  float*  wgt  = (float*) alloc((size_t)M_ * 3 * sizeof(float));
  float*  psum = (float*) alloc((size_t)256 * NOUT_ * sizeof(float));
  float*  psq  = (float*) alloc((size_t)256 * NOUT_ * sizeof(float));
  float*  mu   = (float*) alloc(NOUT_ * sizeof(float));
  float*  rsig = (float*) alloc(NOUT_ * sizeof(float));
  half_t* Wh0  = (half_t*)alloc((size_t)NOUT_ * KIN_ * sizeof(half_t));
  half_t* Wh1  = (half_t*)alloc((size_t)NOUT_ * NOUT_ * sizeof(half_t));
  half_t* hA   = (half_t*)alloc((size_t)M_ * KIN_ * sizeof(half_t)); // reused as h1
  float*  z    = (float*) alloc((size_t)M_ * NOUT_ * sizeof(float)); // both layers
  half_t* h1   = hA;   // layer-1 f16 activations alias the (dead) layer-0 input

  // 1) 3-NN + weights
  knn_kernel<<<B_ * (N2_ / 256), 256, 0, stream>>>(pos, prev_pos, idx, wgt);

  // 2) interpolate + concat -> f16 A matrix [M, 384]
  build_h_kernel<<<4096, 256, 0, stream>>>(x, prev_x, idx, wgt, hA);

  // 3) weights -> f16
  cvt_f16_kernel<<<(NOUT_ * KIN_ / 4 + 255) / 256, 256, 0, stream>>>(
      W0, Wh0, NOUT_ * KIN_ / 4);
  cvt_f16_kernel<<<(NOUT_ * NOUT_ / 4 + 255) / 256, 256, 0, stream>>>(
      W1, Wh1, NOUT_ * NOUT_ / 4);

  // 4) layer 0: GEMM + bias, deterministic BN stats, norm+relu -> f16
  gemm_kernel<<<dim3(M_ / BM, NOUT_ / BN), 256, 0, stream>>>(hA, Wh0, b0, z, KIN_);
  colsum_kernel<<<M_ / 256, 64, 0, stream>>>(z, psum, psq);
  bn_stats_kernel<<<1, NOUT_, 0, stream>>>(psum, psq, mu, rsig);
  norm_relu_f16_kernel<<<4096, 256, 0, stream>>>(z, mu, rsig, g0, be0, h1);

  // 5) layer 1: GEMM + bias, BN stats, norm+relu -> f32 output
  gemm_kernel<<<dim3(M_ / BM, NOUT_ / BN), 256, 0, stream>>>(h1, Wh1, b1, z, NOUT_);
  colsum_kernel<<<M_ / 256, 64, 0, stream>>>(z, psum, psq);
  bn_stats_kernel<<<1, NOUT_, 0, stream>>>(psum, psq, mu, rsig);
  norm_relu_f32_kernel<<<4096, 256, 0, stream>>>(z, mu, rsig, g1, be1, out);

  // 6) second tuple element: prev_pos passthrough
  copy_f32x4_kernel<<<(B_ * N2_ * 3 / 4 + 255) / 256, 256, 0, stream>>>(
      prev_pos, out + (size_t)M_ * NOUT_, B_ * N2_ * 3 / 4);
}
